// BertForSentenceClassification_60318520705309
// MI455X (gfx1250) — matrix operand order repaired and checked by reading, not compiled, not run on hardware
//
#include <hip/hip_runtime.h>
#include <hip/hip_bf16.h>

// Problem constants (match reference)
#define BB   64
#define LL   1024
#define HH   768
#define SS   32
#define NHH  8
#define HDD  96
#define SEPID 102
#define MM   (BB*SS)   // 2048 rows through the MLP stack

typedef __attribute__((ext_vector_type(16))) _Float16 v16h;
typedef __attribute__((ext_vector_type(8)))  _Float16 v8h;
typedef __attribute__((ext_vector_type(8)))  float    v8f;

// ---------------------------------------------------------------- sep finder
__global__ void k_find_seps(const int* __restrict__ ids,
                            int* __restrict__ starts, int* __restrict__ ends) {
  int b = threadIdx.x;
  if (b >= BB) return;
  int sep[SS];
  int cnt = 0;
  for (int t = 0; t < LL; ++t) {
    int v = ids[b * LL + t];
    if (v == SEPID && cnt < SS) sep[cnt++] = t;
  }
  for (int i = cnt; i < SS; ++i) sep[i] = LL - 1;
  for (int s = 0; s < SS; ++s) {
    starts[b * SS + s] = (s == 0) ? 0 : sep[s - 1];
    ends[b * SS + s]   = sep[s];
  }
}

// ---------------------------------------------------------------- f32 -> f16
__global__ void k_f32_to_f16(const float* __restrict__ in, _Float16* __restrict__ out, int n) {
  int i = blockIdx.x * blockDim.x + threadIdx.x;
  if (i < n) out[i] = (_Float16)in[i];
}

// ------------------------------------------------------- segment mean pooling
// grid = B*S blocks, 256 threads; memory-bound stage (streams sequence_output)
__global__ void k_seg_pool(const float* __restrict__ seq, const int* __restrict__ am,
                           const int* __restrict__ starts, const int* __restrict__ ends,
                           _Float16* __restrict__ reps_h) {
  int bs = blockIdx.x;
  int b  = bs / SS;
  int st = starts[bs], en = ends[bs];
  float denom = 1e-10f;
  for (int t = st; t <= en; ++t) denom += (float)am[b * LL + t];
  float inv = 1.0f / denom;
  for (int h = threadIdx.x; h < HH; h += blockDim.x) {
    float acc = 0.f;
    for (int t = st; t <= en; ++t) {
      float w = (float)am[b * LL + t];
      acc += w * seq[((size_t)b * LL + t) * HH + h];
    }
    reps_h[(size_t)bs * HH + h] = (_Float16)(acc * inv);
  }
}

// ------------------------------------------------------------ WMMA fragments
// Contiguous k order chosen here, same for A and B (16-bit 16x32 A, 32x16 B, wave32).
// A: lane l -> row M=l%16; half j -> K = (l>=16?16:0) + j
// B (for X·W^T, B[k][n]=W[n][k]): lane l -> col N=l%16; half j -> K=(l>=16?16:0)+j
__device__ inline v16h frag_a(const _Float16* __restrict__ row, int k0, int offA) {
  v8h a0 = *(const v8h*)(row + k0 + offA);
  v8h a1 = *(const v8h*)(row + k0 + 8 + offA);
  v16h A;
#pragma unroll
  for (int j = 0; j < 8; ++j) { A[j] = a0[j]; A[8 + j] = a1[j]; }
  return A;
}
__device__ inline v16h frag_b(const _Float16* __restrict__ row, int k0, int offB) {
  // contiguous 16 halves, 32B aligned -> single vector load
  return *(const v16h*)(row + k0 + offB);
}

// ------------------------------------------------------------ WMMA GEMM core
// Y[M,N] = X[M,K] @ W[N,K]^T + bias ; optional f16 copy Yh.
// One wave = 32x32 output tile (2x2 grid of 16x16 WMMA tiles): 4 independent
// accumulator chains per K-step -> back-to-back v_wmma issue, 16 FLOP/byte of
// operand traffic. 4 waves/block tile 128 columns.
__global__ void k_gemm_wmma_bias(const _Float16* __restrict__ X,
                                 const _Float16* __restrict__ W,
                                 const float* __restrict__ bias,
                                 float* __restrict__ Y,
                                 _Float16* __restrict__ Yh,
                                 int M, int N, int K) {
  int lane = threadIdx.x & 31;
  int wave = threadIdx.x >> 5;
  int n0 = (blockIdx.x * 4 + wave) * 32;
  int m0 = blockIdx.y * 32;
  if (n0 >= N || m0 >= M) return;          // wave-uniform exit (EXEC stays full)

  int rsel = lane & 15;
  int hi   = lane >> 4;                    // 0: lanes 0-15, 1: lanes 16-31
  const _Float16* xrow0 = X + (size_t)(m0 + rsel) * K;
  const _Float16* xrow1 = X + (size_t)(m0 + 16 + rsel) * K;
  const _Float16* wrow0 = W + (size_t)(n0 + rsel) * K;
  const _Float16* wrow1 = W + (size_t)(n0 + 16 + rsel) * K;
  int offA = hi * 16;
  int offB = hi * 16;

  v8f acc00 = {}, acc01 = {}, acc10 = {}, acc11 = {};
  for (int k0 = 0; k0 < K; k0 += 32) {
    v16h A0 = frag_a(xrow0, k0, offA);
    v16h A1 = frag_a(xrow1, k0, offA);
    v16h B0 = frag_b(wrow0, k0, offB);
    v16h B1 = frag_b(wrow1, k0, offB);
    acc00 = __builtin_amdgcn_wmma_f32_16x16x32_f16(false, A0, false, B0, (short)0, acc00, false, false);
    acc01 = __builtin_amdgcn_wmma_f32_16x16x32_f16(false, A0, false, B1, (short)0, acc01, false, false);
    acc10 = __builtin_amdgcn_wmma_f32_16x16x32_f16(false, A1, false, B0, (short)0, acc10, false, false);
    acc11 = __builtin_amdgcn_wmma_f32_16x16x32_f16(false, A1, false, B1, (short)0, acc11, false, false);
  }

  // D layout: lane -> N=l%16, vgpr i -> M = i + (l>=16?8:0)
  int nA = n0 + rsel;
  int nB = n0 + 16 + rsel;
  float bvA = bias ? bias[nA] : 0.f;
  float bvB = bias ? bias[nB] : 0.f;
#pragma unroll
  for (int i = 0; i < 8; ++i) {
    int mA = m0 + i + hi * 8;
    int mB = m0 + 16 + i + hi * 8;
    float v00 = acc00[i] + bvA;
    float v01 = acc01[i] + bvB;
    float v10 = acc10[i] + bvA;
    float v11 = acc11[i] + bvB;
    Y[(size_t)mA * N + nA] = v00;
    Y[(size_t)mA * N + nB] = v01;
    Y[(size_t)mB * N + nA] = v10;
    Y[(size_t)mB * N + nB] = v11;
    if (Yh) {
      Yh[(size_t)mA * N + nA] = (_Float16)v00;
      Yh[(size_t)mA * N + nB] = (_Float16)v01;
      Yh[(size_t)mB * N + nA] = (_Float16)v10;
      Yh[(size_t)mB * N + nB] = (_Float16)v11;
    }
  }
}

// ----------------------------------------------------------- LayerNorm + GELU
// grid = rows (2048), 256 threads, 768 cols (3 per thread)
__global__ void k_ln_gelu(const float* __restrict__ in, const float* __restrict__ g,
                          const float* __restrict__ bta, float* __restrict__ o32,
                          _Float16* __restrict__ o16) {
  __shared__ float red[256];
  int row = blockIdx.x;
  const float* x = in + (size_t)row * HH;
  float v[3];
  float s = 0.f;
#pragma unroll
  for (int r = 0; r < 3; ++r) { v[r] = x[threadIdx.x + r * 256]; s += v[r]; }
  red[threadIdx.x] = s; __syncthreads();
  for (int off = 128; off > 0; off >>= 1) {
    if (threadIdx.x < off) red[threadIdx.x] += red[threadIdx.x + off];
    __syncthreads();
  }
  float mean = red[0] / (float)HH;
  __syncthreads();
  s = 0.f;
#pragma unroll
  for (int r = 0; r < 3; ++r) { float d = v[r] - mean; s += d * d; }
  red[threadIdx.x] = s; __syncthreads();
  for (int off = 128; off > 0; off >>= 1) {
    if (threadIdx.x < off) red[threadIdx.x] += red[threadIdx.x + off];
    __syncthreads();
  }
  float rstd = rsqrtf(red[0] / (float)HH + 1e-5f);
#pragma unroll
  for (int r = 0; r < 3; ++r) {
    int h = threadIdx.x + r * 256;
    float y = (v[r] - mean) * rstd * g[h] + bta[h];
    float z = 0.5f * y * (1.0f + erff(y * 0.70710678118654752f));
    if (o32) o32[(size_t)row * HH + h] = z;
    if (o16) o16[(size_t)row * HH + h] = (_Float16)z;
  }
}

// ---------------------------------------------------------------- attention
// grid = B*NH blocks, 128 threads; S=32 tokens, HD=96 per head (tiny).
__global__ void k_attention(const float* __restrict__ qkv, _Float16* __restrict__ o_h) {
  __shared__ float qs[SS * HDD], ks[SS * HDD], vs[SS * HDD];
  __shared__ float sc[SS * SS];
  int b  = blockIdx.x / NHH;
  int hh = blockIdx.x % NHH;
  int tid = threadIdx.x;

  for (int i = tid; i < SS * HDD; i += 128) {
    int s = i / HDD, d = i % HDD;
    size_t base = ((size_t)(b * SS + s)) * (3 * HH) + hh * HDD + d;
    qs[i] = qkv[base];
    ks[i] = qkv[base + HH];
    vs[i] = qkv[base + 2 * HH];
  }
  __syncthreads();

  const float scale = rsqrtf((float)HDD);
  for (int i = tid; i < SS * SS; i += 128) {
    int q = i / SS, k = i % SS;
    float dot = 0.f;
    for (int d = 0; d < HDD; ++d) dot += qs[q * HDD + d] * ks[k * HDD + d];
    sc[i] = dot * scale;
  }
  __syncthreads();

  if (tid < SS) {                          // row-wise softmax
    float mx = -3.4e38f;
    for (int k = 0; k < SS; ++k) mx = fmaxf(mx, sc[tid * SS + k]);
    float sum = 0.f;
    for (int k = 0; k < SS; ++k) { float e = expf(sc[tid * SS + k] - mx); sc[tid * SS + k] = e; sum += e; }
    float inv = 1.0f / sum;
    for (int k = 0; k < SS; ++k) sc[tid * SS + k] *= inv;
  }
  __syncthreads();

  for (int i = tid; i < SS * HDD; i += 128) {
    int q = i / HDD, d = i % HDD;
    float acc = 0.f;
    for (int k = 0; k < SS; ++k) acc += sc[q * SS + k] * vs[k * HDD + d];
    o_h[((size_t)(b * SS + q)) * HH + hh * HDD + d] = (_Float16)acc;
  }
}

// ------------------------------------------------------------------- logits
// grid = 2048 rows, 128 threads; 2-class projection in f32
__global__ void k_logits(const float* __restrict__ hf, const float* __restrict__ W2,
                         const float* __restrict__ b2, float* __restrict__ outp) {
  __shared__ float s0[128], s1[128];
  int row = blockIdx.x;
  const float* x = hf + (size_t)row * HH;
  float p0 = 0.f, p1 = 0.f;
  for (int k = threadIdx.x; k < HH; k += 128) {
    float xv = x[k];
    p0 += xv * W2[k];
    p1 += xv * W2[HH + k];
  }
  s0[threadIdx.x] = p0; s1[threadIdx.x] = p1; __syncthreads();
  for (int off = 64; off > 0; off >>= 1) {
    if (threadIdx.x < off) { s0[threadIdx.x] += s0[threadIdx.x + off]; s1[threadIdx.x] += s1[threadIdx.x + off]; }
    __syncthreads();
  }
  if (threadIdx.x == 0) {
    outp[1 + row * 2 + 0] = s0[0] + b2[0];
    outp[1 + row * 2 + 1] = s1[0] + b2[1];
  }
}

// --------------------------------------------------------------------- loss
__global__ void k_loss(const float* __restrict__ outp, const int* __restrict__ labels,
                       float* __restrict__ loss_out) {
  __shared__ float sred[256];
  __shared__ int   cred[256];
  float lsum = 0.f; int lcnt = 0;
  for (int i = threadIdx.x; i < BB * SS; i += 256) {
    int lab = labels[i];
    if (lab != -100) {
      float a = outp[1 + i * 2], b = outp[1 + i * 2 + 1];
      float mx = fmaxf(a, b);
      float lse = mx + logf(expf(a - mx) + expf(b - mx));
      float sel = (lab == 0) ? a : b;
      lsum += lse - sel;
      lcnt += 1;
    }
  }
  sred[threadIdx.x] = lsum; cred[threadIdx.x] = lcnt; __syncthreads();
  for (int off = 128; off > 0; off >>= 1) {
    if (threadIdx.x < off) { sred[threadIdx.x] += sred[threadIdx.x + off]; cred[threadIdx.x] += cred[threadIdx.x + off]; }
    __syncthreads();
  }
  if (threadIdx.x == 0) loss_out[0] = sred[0] / (float)((cred[0] > 0) ? cred[0] : 1);
}

// ---------------------------------------------------------------------------
extern "C" void kernel_launch(void* const* d_in, const int* in_sizes, int n_in,
                              void* d_out, int out_size, void* d_ws, size_t ws_size,
                              hipStream_t stream) {
  (void)in_sizes; (void)n_in; (void)out_size; (void)ws_size;

  const int*   ids    = (const int*)d_in[0];
  const int*   amask  = (const int*)d_in[1];
  const int*   labels = (const int*)d_in[2];
  const float* seq    = (const float*)d_in[3];
  const float* enc_W  = (const float*)d_in[4];
  const float* enc_b  = (const float*)d_in[5];
  const float* enc_g  = (const float*)d_in[6];
  const float* enc_bt = (const float*)d_in[7];
  const float* inW    = (const float*)d_in[8];
  const float* inB    = (const float*)d_in[9];
  const float* outW   = (const float*)d_in[10];
  const float* outB   = (const float*)d_in[11];
  const float* c1W    = (const float*)d_in[12];
  const float* c1B    = (const float*)d_in[13];
  const float* c1g    = (const float*)d_in[14];
  const float* c1bt   = (const float*)d_in[15];
  const float* c2W    = (const float*)d_in[16];
  const float* c2B    = (const float*)d_in[17];
  float* outp = (float*)d_out;

  // workspace carve-up (256B aligned)
  char* base = (char*)d_ws;
  size_t off = 0;
  auto carve = [&](size_t bytes) -> char* {
    char* p = base + off;
    off = (off + bytes + 255) & ~(size_t)255;
    return p;
  };
  int*       starts = (int*)      carve((size_t)BB * SS * 4);
  int*       ends   = (int*)      carve((size_t)BB * SS * 4);
  _Float16*  encWh  = (_Float16*) carve((size_t)HH * HH * 2);
  _Float16*  inWh   = (_Float16*) carve((size_t)3 * HH * HH * 2);
  _Float16*  outWh  = (_Float16*) carve((size_t)HH * HH * 2);
  _Float16*  c1Wh   = (_Float16*) carve((size_t)HH * HH * 2);
  _Float16*  repsh  = (_Float16*) carve((size_t)MM * HH * 2);
  float*     tmp1   = (float*)    carve((size_t)MM * HH * 4);
  _Float16*  xh     = (_Float16*) carve((size_t)MM * HH * 2);
  float*     qkv    = (float*)    carve((size_t)MM * 3 * HH * 4);
  _Float16*  oh     = (_Float16*) carve((size_t)MM * HH * 2);
  float*     att    = (float*)    carve((size_t)MM * HH * 4);
  _Float16*  atth   = (_Float16*) carve((size_t)MM * HH * 2);
  float*     tmp2   = (float*)    carve((size_t)MM * HH * 4);
  float*     hfeat  = (float*)    carve((size_t)MM * HH * 4);

  // 1) sep positions -> segment bounds
  k_find_seps<<<1, 64, 0, stream>>>(ids, starts, ends);

  // 2) weights to f16 (WMMA operand precision)
  {
    int n1 = HH * HH, n3 = 3 * HH * HH;
    k_f32_to_f16<<<(n1 + 255) / 256, 256, 0, stream>>>(enc_W, encWh, n1);
    k_f32_to_f16<<<(n3 + 255) / 256, 256, 0, stream>>>(inW,   inWh,  n3);
    k_f32_to_f16<<<(n1 + 255) / 256, 256, 0, stream>>>(outW,  outWh, n1);
    k_f32_to_f16<<<(n1 + 255) / 256, 256, 0, stream>>>(c1W,   c1Wh,  n1);
  }

  // 3) segment mean pooling (memory-bound, streams 200MB once)
  k_seg_pool<<<BB * SS, 256, 0, stream>>>(seq, amask, starts, ends, repsh);

  // GEMM grids: block = 4 waves x (32x32 tile) = 32 rows x 128 cols
  dim3 gemmBlk(128);
  dim3 g768(HH / 128, MM / 32);        // (6, 64)
  dim3 g2304(3 * HH / 128, MM / 32);   // (18, 64)

  // 4) enc GEMM: tmp1 = reps @ enc_W^T + enc_b
  k_gemm_wmma_bias<<<g768, gemmBlk, 0, stream>>>(
      repsh, encWh, enc_b, tmp1, (_Float16*)nullptr, MM, HH, HH);

  // 5) x = gelu(LN(tmp1)) -> f16
  k_ln_gelu<<<MM, 256, 0, stream>>>(tmp1, enc_g, enc_bt, (float*)nullptr, xh);

  // 6) qkv GEMM: qkv = x @ in_proj_W^T + b (N = 2304)
  k_gemm_wmma_bias<<<g2304, gemmBlk, 0, stream>>>(
      xh, inWh, inB, qkv, (_Float16*)nullptr, MM, 3 * HH, HH);

  // 7) per-(batch,head) attention -> o (f16)
  k_attention<<<BB * NHH, 128, 0, stream>>>(qkv, oh);

  // 8) out_proj GEMM: att = o @ out_proj_W^T + b (keep f16 copy for next GEMM)
  k_gemm_wmma_bias<<<g768, gemmBlk, 0, stream>>>(
      oh, outWh, outB, att, atth, MM, HH, HH);

  // 9) cls GEMM: tmp2 = att @ cls_W1^T + b1
  k_gemm_wmma_bias<<<g768, gemmBlk, 0, stream>>>(
      atth, c1Wh, c1B, tmp2, (_Float16*)nullptr, MM, HH, HH);

  // 10) h = gelu(LN(tmp2)) -> f32
  k_ln_gelu<<<MM, 256, 0, stream>>>(tmp2, c1g, c1bt, hfeat, (_Float16*)nullptr);

  // 11) logits (2 classes, f32) -> d_out[1..4096]
  k_logits<<<MM, 128, 0, stream>>>(hfeat, c2W, c2B, outp);

  // 12) masked mean NLL -> d_out[0]
  k_loss<<<1, 256, 0, stream>>>(outp, labels, outp);
}
